// MultiHeadSelfAttention_48799418417707
// MI455X (gfx1250) — matrix operand order repaired and checked by reading, not compiled, not run on hardware
//
#include <hip/hip_runtime.h>
#include <hip/hip_bf16.h>

// ---------------------------------------------------------------------------
// Multi-head causal self-attention, CDNA5 (gfx1250) WMMA bf16 pipeline.
//   B=4, T=2048, C=1024, H=16, hs=64
// Stages (bf16 compute, fp32 accumulate):
//   1a. convert x fp32 -> bf16
//   1b. convert + transpose W_qkv/W_proj fp32 -> bf16  (Wt[N][K])
//   2.  QKV GEMM: TDM (tensor_load_to_lds) double-buffered tiles, WMMA
//       -> q(1/8-scaled), k in [B,H,T,hs], v transposed into [B,H,hs,T]
//   3.  flash attention: one wave per 16-query tile, online softmax
//   4.  proj GEMM: same TDM+WMMA scheme, + bias -> fp32 output
// ---------------------------------------------------------------------------

#define B_   4
#define T_   2048
#define C_   1024
#define H_   16
#define HS_  64
#define N3_  3072   // 3*C

typedef __attribute__((ext_vector_type(16))) __bf16    v16bf;
typedef __attribute__((ext_vector_type(8)))  float     v8f;
typedef __attribute__((ext_vector_type(4)))  unsigned  v4u;
typedef __attribute__((ext_vector_type(8)))  int       v8i;
typedef __attribute__((ext_vector_type(4)))  int       v4i;

union Frag16 { v16bf v; unsigned short s[16]; };

__device__ __forceinline__ unsigned short f2bf(float f) {
    unsigned u = __float_as_uint(f);
    u += 0x7FFFu + ((u >> 16) & 1u);          // round-to-nearest-even
    return (unsigned short)(u >> 16);
}

// A-matrix (16x32 bf16) element->K map: K = hi*16 + j (matches B).
__device__ __forceinline__ int a_kidx(int j, int hi) {
    return hi * 16 + j;
}
// B-matrix (32x16 bf16): lane lo = N column, element j -> K = hi*16 + j.

__device__ __forceinline__ v8f wmma_bf16(const Frag16& a, const Frag16& b, v8f c) {
    return __builtin_amdgcn_wmma_f32_16x16x32_bf16(
        false, a.v, false, b.v, (short)0, c, false, false);
}

// ---------------------------------------------------------------------------
// Tensor Data Mover: 2D bf16 tile (tile_d0 elems/row x tile_d1 rows) from
// global (row stride = stride_elems) into contiguous LDS.  D# per ISA §8.
// ---------------------------------------------------------------------------
__device__ __forceinline__ void tdm_load_tile_2d(
    unsigned lds_off, const void* gptr,
    unsigned tensor_d0, unsigned tensor_d1, unsigned stride_elems,
    unsigned tile_d0, unsigned tile_d1) {
    unsigned long long ga = (unsigned long long)(size_t)gptr;
    v4u g0;
    g0.x = 1u;                                        // count=1 (valid, user)
    g0.y = lds_off;                                   // LDS byte address
    g0.z = (unsigned)(ga & 0xFFFFFFFFu);              // global_addr[31:0]
    g0.w = (unsigned)((ga >> 32) & 0x01FFFFFFu)       // global_addr[56:32]
         | (2u << 30);                                // type=2 ("image")
    v8i g1;
    g1[0] = (int)(1u << 16);                          // data_size=1 (2 bytes)
    g1[1] = (int)((tensor_d0 & 0xFFFFu) << 16);       // tensor_dim0[15:0]
    g1[2] = (int)(((tensor_d0 >> 16) & 0xFFFFu) |     // tensor_dim0[31:16]
                  ((tensor_d1 & 0xFFFFu) << 16));     // tensor_dim1[15:0]
    g1[3] = (int)(((tensor_d1 >> 16) & 0xFFFFu) |     // tensor_dim1[31:16]
                  ((tile_d0 & 0xFFFFu) << 16));       // tile_dim0
    g1[4] = (int)(tile_d1 & 0xFFFFu);                 // tile_dim1 (tile_dim2=0)
    g1[5] = (int)stride_elems;                        // tensor_dim0_stride[31:0]
    g1[6] = 0;                                        // stride[47:32], dim1_stride lo
    g1[7] = 0;
    v4i z = {0, 0, 0, 0};
#if defined(__clang_major__) && __clang_major__ >= 23
    v8i z8 = {0, 0, 0, 0, 0, 0, 0, 0};
    __builtin_amdgcn_tensor_load_to_lds(g0, g1, z, z, z8, 0);
#else
    __builtin_amdgcn_tensor_load_to_lds(g0, g1, z, z, 0);
#endif
}
__device__ __forceinline__ void wait_tensor0() {
    __builtin_amdgcn_s_wait_tensorcnt(0);
}

// ---------------------------------------------------------------------------
// Stage 1a: fp32 -> bf16 (x)
// ---------------------------------------------------------------------------
__global__ void cvt_f32_to_bf16(const float* __restrict__ src,
                                unsigned short* __restrict__ dst, int n) {
    for (int i = blockIdx.x * blockDim.x + threadIdx.x; i < n;
         i += gridDim.x * blockDim.x)
        dst[i] = f2bf(src[i]);
}

// ---------------------------------------------------------------------------
// Stage 1b: fp32 [R,Ccols] -> bf16 transposed [Ccols,R]  (LDS-tiled)
// ---------------------------------------------------------------------------
__global__ __launch_bounds__(256) void cvt_transpose_bf16(
    const float* __restrict__ src, unsigned short* __restrict__ dst,
    int R, int Ccols) {
    __shared__ float tile[32][33];
    const int c0 = blockIdx.x * 32, r0 = blockIdx.y * 32;
    const int tx = threadIdx.x, ty = threadIdx.y;   // 32 x 8
#pragma unroll
    for (int dy = 0; dy < 32; dy += 8)
        tile[ty + dy][tx] = src[(size_t)(r0 + ty + dy) * Ccols + c0 + tx];
    __syncthreads();
#pragma unroll
    for (int dy = 0; dy < 32; dy += 8)
        dst[(size_t)(c0 + ty + dy) * R + r0 + tx] = f2bf(tile[tx][ty + dy]);
}

// ---------------------------------------------------------------------------
// Shared GEMM core: block tile 128(M) x 64(N), K-steps of 32, 4 waves,
// each wave computes 32x64.  TDM double-buffered LDS staging.
// acc[h2][c] indexed by M-half and N-chunk.
// ---------------------------------------------------------------------------
#define GEMM_CORE(Aptr, Bptr)                                                  \
    __shared__ unsigned short As[2][128][32];                                  \
    __shared__ unsigned short Bs[2][64][32];                                   \
    const int tid  = threadIdx.x;                                              \
    const int wave = tid >> 5;                                                 \
    const int lane = tid & 31;                                                 \
    const int lo   = lane & 15;                                                \
    const int hi   = lane >> 4;                                                \
    const int m0   = blockIdx.x * 128;                                         \
    const int n0   = blockIdx.y * 64;                                          \
    v8f acc[2][4];                                                             \
    _Pragma("unroll") for (int h2 = 0; h2 < 2; ++h2)                           \
        _Pragma("unroll") for (int c = 0; c < 4; ++c)                          \
            acc[h2][c] = (v8f){0,0,0,0,0,0,0,0};                               \
    if (wave == 0) {                                                           \
        tdm_load_tile_2d((unsigned)(size_t)&As[0][0][0],                       \
                         (Aptr) + (size_t)m0 * C_, C_, 0x100000u, C_, 32, 128);\
        tdm_load_tile_2d((unsigned)(size_t)&Bs[0][0][0],                       \
                         (Bptr) + (size_t)n0 * C_, C_, 0x100000u, C_, 32, 64); \
        wait_tensor0();                                                        \
    }                                                                          \
    __syncthreads();                                                           \
    for (int k0 = 0; k0 < C_; k0 += 32) {                                      \
        const int buf = (k0 >> 5) & 1;                                         \
        if (wave == 0 && k0 + 32 < C_) {                                       \
            tdm_load_tile_2d((unsigned)(size_t)&As[buf ^ 1][0][0],             \
                             (Aptr) + (size_t)m0 * C_ + k0 + 32,               \
                             C_, 0x100000u, C_, 32, 128);                      \
            tdm_load_tile_2d((unsigned)(size_t)&Bs[buf ^ 1][0][0],             \
                             (Bptr) + (size_t)n0 * C_ + k0 + 32,               \
                             C_, 0x100000u, C_, 32, 64);                       \
        }                                                                      \
        Frag16 af[2];                                                          \
        _Pragma("unroll") for (int h2 = 0; h2 < 2; ++h2)                       \
            _Pragma("unroll") for (int j = 0; j < 16; ++j)                     \
                af[h2].s[j] = As[buf][32 * wave + h2 * 16 + lo][a_kidx(j, hi)];\
        _Pragma("unroll") for (int c = 0; c < 4; ++c) {                        \
            Frag16 bf;                                                         \
            _Pragma("unroll") for (int j = 0; j < 16; ++j)                     \
                bf.s[j] = Bs[buf][c * 16 + lo][hi * 16 + j];                   \
            acc[0][c] = wmma_bf16(af[0], bf, acc[0][c]);                       \
            acc[1][c] = wmma_bf16(af[1], bf, acc[1][c]);                       \
        }                                                                      \
        if (wave == 0) wait_tensor0();                                         \
        __syncthreads();                                                       \
    }

// ---------------------------------------------------------------------------
// Stage 2: QKV GEMM  [8192,1024] x Wt[3072,1024].
// Scatters per-head q/k slices into [B,H,T,hs], v into [B,H,hs,T].
// ---------------------------------------------------------------------------
__global__ __launch_bounds__(128) void qkv_gemm_kernel(
    const unsigned short* __restrict__ Xb,    // [8192,1024] bf16
    const unsigned short* __restrict__ Wt,    // [3072,1024] bf16 (transposed)
    const float* __restrict__ bias,           // [3072]
    unsigned short* __restrict__ q,           // [B,H,T,hs]
    unsigned short* __restrict__ k,           // [B,H,T,hs]
    unsigned short* __restrict__ vT) {        // [B,H,hs,T]
    GEMM_CORE(Xb, Wt)

#pragma unroll
    for (int h2 = 0; h2 < 2; ++h2) {
#pragma unroll
        for (int c = 0; c < 4; ++c) {
#pragma unroll
            for (int r = 0; r < 8; ++r) {
                int row = r + hi * 8;
                int gm  = m0 + wave * 32 + h2 * 16 + row;
                int gn  = n0 + c * 16 + lo;
                float val = acc[h2][c][r] + bias[gn];
                int h     = gn / 192;
                int rem   = gn - h * 192;
                int which = rem >> 6;
                int d     = rem & 63;
                int bidx  = gm >> 11;           // / T_
                int t     = gm & (T_ - 1);
                int bh    = bidx * H_ + h;
                if (which == 2) {
                    vT[((size_t)bh * HS_ + d) * T_ + t] = f2bf(val);
                } else {
                    size_t o = ((size_t)bh * T_ + t) * HS_ + d;
                    if (which == 0) q[o] = f2bf(val * 0.125f);
                    else            k[o] = f2bf(val);
                }
            }
        }
    }
}

// ---------------------------------------------------------------------------
// Stage 4: proj GEMM [8192,1024] x Wt[1024,1024] + bias -> fp32
// ---------------------------------------------------------------------------
__global__ __launch_bounds__(128) void proj_gemm_kernel(
    const unsigned short* __restrict__ Ab,    // [8192,1024] bf16
    const unsigned short* __restrict__ Wt,    // [1024,1024] bf16 (transposed)
    const float* __restrict__ bias,           // [1024]
    float* __restrict__ out) {                // [8192,1024]
    GEMM_CORE(Ab, Wt)

#pragma unroll
    for (int h2 = 0; h2 < 2; ++h2) {
#pragma unroll
        for (int c = 0; c < 4; ++c) {
#pragma unroll
            for (int r = 0; r < 8; ++r) {
                int row = r + hi * 8;
                int gm  = m0 + wave * 32 + h2 * 16 + row;
                int gn  = n0 + c * 16 + lo;
                out[(size_t)gm * C_ + gn] = acc[h2][c][r] + bias[gn];
            }
        }
    }
}

// ---------------------------------------------------------------------------
// Stage 3: flash attention. One wave per (b,h, 16-query tile); 32 keys/iter.
// ---------------------------------------------------------------------------
__global__ __launch_bounds__(128) void flash_attn_kernel(
    const unsigned short* __restrict__ q,   // [B,H,T,hs] bf16, pre-scaled
    const unsigned short* __restrict__ k,   // [B,H,T,hs]
    const unsigned short* __restrict__ vT,  // [B,H,hs,T]
    unsigned short* __restrict__ out) {     // [B,T,C] bf16
    __shared__ unsigned short Plds[4][16][32];

    const int tid  = threadIdx.x;
    const int wave = tid >> 5;
    const int lane = tid & 31;
    const int lo   = lane & 15;
    const int hi   = lane >> 4;

    const int qtile = blockIdx.x * 4 + wave;        // 0 .. B*H*(T/16)-1
    const int bh    = qtile >> 7;                   // / (T/16)
    const int q0    = (qtile & 127) * 16;
    const size_t kbase = (size_t)bh * T_ * HS_;
    const size_t vbase = (size_t)bh * HS_ * T_;

    // Q tile 16x64 as two A fragments (K = 0..31 and 32..63)
    Frag16 qa0, qa1;
#pragma unroll
    for (int j = 0; j < 16; ++j) {
        int d = a_kidx(j, hi);
        qa0.s[j] = q[kbase + (size_t)(q0 + lo) * HS_ + d];
        qa1.s[j] = q[kbase + (size_t)(q0 + lo) * HS_ + 32 + d];
    }

    v8f o0 = (v8f){0,0,0,0,0,0,0,0}, o1 = o0, o2 = o0, o3 = o0;
    float mrow[8], lrow[8];
#pragma unroll
    for (int r = 0; r < 8; ++r) { mrow[r] = -3.0e38f; lrow[r] = 0.0f; }

    for (int kb = 0; kb < q0 + 16; kb += 32) {
        if (kb + 32 < q0 + 16)
            __builtin_prefetch(&k[kbase + (size_t)(kb + 32 + lane) * HS_], 0, 3);

        // ---- S = Q K^T for keys [kb, kb+32) : two 16x16 tiles ----
        v8f S0 = (v8f){0,0,0,0,0,0,0,0}, S1 = S0;
#pragma unroll
        for (int st = 0; st < 2; ++st) {
            int key = kb + st * 16 + lo;            // B-frag: lane = key col
#pragma unroll
            for (int kc = 0; kc < 2; ++kc) {        // head-dim halves
                Frag16 f;
                const unsigned short* kp =
                    &k[kbase + (size_t)key * HS_ + kc * 32 + hi * 16];
                *(uint4*)&f.s[0] = *(const uint4*)kp;
                *(uint4*)&f.s[8] = *(const uint4*)(kp + 8);
                if (st == 0) S0 = wmma_bf16(kc ? qa1 : qa0, f, S0);
                else         S1 = wmma_bf16(kc ? qa1 : qa0, f, S1);
            }
        }

        // ---- causal mask + online softmax ----
#pragma unroll
        for (int r = 0; r < 8; ++r) {
            int qrow = q0 + r + hi * 8;
            if (kb + lo      > qrow) S0[r] = -1.0e9f;
            if (kb + 16 + lo > qrow) S1[r] = -1.0e9f;

            float mx = fmaxf(S0[r], S1[r]);
#pragma unroll
            for (int off = 1; off < 16; off <<= 1)
                mx = fmaxf(mx, __shfl_xor(mx, off, 16));
            float mnew = fmaxf(mrow[r], mx);
            float corr = __expf(mrow[r] - mnew);
            float p0 = __expf(S0[r] - mnew);
            float p1 = __expf(S1[r] - mnew);
            float rs = p0 + p1;
#pragma unroll
            for (int off = 1; off < 16; off <<= 1)
                rs += __shfl_xor(rs, off, 16);
            lrow[r] = lrow[r] * corr + rs;
            mrow[r] = mnew;
            o0[r] *= corr; o1[r] *= corr; o2[r] *= corr; o3[r] *= corr;

            Plds[wave][r + hi * 8][lo]      = f2bf(p0);
            Plds[wave][r + hi * 8][16 + lo] = f2bf(p1);
        }
        asm volatile("s_wait_dscnt 0x0" ::: "memory");
        __builtin_amdgcn_wave_barrier();

        // ---- P (16x32) as A fragment (two contiguous b128 runs) ----
        Frag16 pf;
#pragma unroll
        for (int j = 0; j < 16; ++j)
            pf.s[j] = Plds[wave][lo][a_kidx(j, hi)];
        __builtin_amdgcn_wave_barrier();

        // ---- O += P V : vT rows are contiguous over keys ----
#pragma unroll
        for (int c = 0; c < 4; ++c) {
            Frag16 vf;
            const unsigned short* vp =
                &vT[vbase + (size_t)(c * 16 + lo) * T_ + kb + hi * 16];
            *(uint4*)&vf.s[0] = *(const uint4*)vp;
            *(uint4*)&vf.s[8] = *(const uint4*)(vp + 8);
            if      (c == 0) o0 = wmma_bf16(pf, vf, o0);
            else if (c == 1) o1 = wmma_bf16(pf, vf, o1);
            else if (c == 2) o2 = wmma_bf16(pf, vf, o2);
            else             o3 = wmma_bf16(pf, vf, o3);
        }
    }

    // ---- normalize + store to [B,T,C] (head-major columns) ----
    const int b = bh / H_, h = bh % H_;
#pragma unroll
    for (int r = 0; r < 8; ++r) {
        int row = r + hi * 8;
        int t   = q0 + row;
        float inv = 1.0f / lrow[r];
        size_t ob = ((size_t)(b * T_ + t)) * C_ + h * HS_ + lo;
        out[ob +  0] = f2bf(o0[r] * inv);
        out[ob + 16] = f2bf(o1[r] * inv);
        out[ob + 32] = f2bf(o2[r] * inv);
        out[ob + 48] = f2bf(o3[r] * inv);
    }
}

// ---------------------------------------------------------------------------
// Host launcher
// ---------------------------------------------------------------------------
extern "C" void kernel_launch(void* const* d_in, const int* in_sizes, int n_in,
                              void* d_out, int out_size, void* d_ws, size_t ws_size,
                              hipStream_t stream) {
    const float* x      = (const float*)d_in[0];   // [4,2048,1024]
    const float* W_qkv  = (const float*)d_in[1];   // [1024,3072]
    const float* b_qkv  = (const float*)d_in[2];   // [3072]
    const float* W_proj = (const float*)d_in[3];   // [1024,1024]
    const float* b_proj = (const float*)d_in[4];   // [1024]
    float* out          = (float*)d_out;           // [4,2048,1024]

    const size_t nX  = (size_t)B_ * T_ * C_;       // 8388608
    const size_t nWq = (size_t)C_ * N3_;           // 3145728
    const size_t nWp = (size_t)C_ * C_;            // 1048576
    const size_t nH  = (size_t)B_ * H_ * T_ * HS_; // 8388608

    unsigned short* ws    = (unsigned short*)d_ws;
    unsigned short* Xb    = ws;            ws += nX;
    unsigned short* Wqt   = ws;            ws += nWq;  // [3072,1024]
    unsigned short* Wpt   = ws;            ws += nWp;  // [1024,1024]
    unsigned short* qb    = ws;            ws += nH;
    unsigned short* kb    = ws;            ws += nH;
    unsigned short* vTb   = ws;            ws += nH;   // [B,H,hs,T]
    unsigned short* attnb = ws;            ws += nX;

    // Stage 1: convert (x) and convert+transpose (weights) to bf16
    cvt_f32_to_bf16<<<2048, 256, 0, stream>>>(x, Xb, (int)nX);
    cvt_transpose_bf16<<<dim3(96, 32), dim3(32, 8), 0, stream>>>(
        W_qkv, Wqt, C_, N3_);
    cvt_transpose_bf16<<<dim3(32, 32), dim3(32, 8), 0, stream>>>(
        W_proj, Wpt, C_, C_);

    // Stage 2: QKV projection (8192/128 x 3072/64 tiles)
    qkv_gemm_kernel<<<dim3(64, 48), 128, 0, stream>>>(Xb, Wqt, b_qkv,
                                                      qb, kb, vTb);

    // Stage 3: flash attention (8192 q-tiles / 4 waves per block)
    flash_attn_kernel<<<2048, 128, 0, stream>>>(qb, kb, vTb, attnb);

    // Stage 4: output projection (8192/128 x 1024/64 tiles)
    proj_gemm_kernel<<<dim3(64, 16), 128, 0, stream>>>(attnb, Wpt, b_proj, out);
}
